// TaskConditionedAttention_54022098649629
// MI455X (gfx1250) — matrix operand order repaired and checked by reading, not compiled, not run on hardware
//
#include <hip/hip_runtime.h>
#include <hip/hip_bf16.h>

// ---------------------------------------------------------------------------
// Task-conditioned MoE attention for MI455X (gfx1250, wave32, WMMA + TDM).
// All heavy matmuls: v_wmma_f32_16x16x32_f16 (f32 accumulate).
// All fragments: b128 loads (global or LDS). A tiles staged by the TDM.
// ---------------------------------------------------------------------------

typedef __attribute__((ext_vector_type(16))) _Float16     v16h;
typedef __attribute__((ext_vector_type(8)))  float        v8f;
typedef __attribute__((ext_vector_type(4)))  unsigned int v4u;
typedef __attribute__((ext_vector_type(8)))  int          v8i;
typedef __attribute__((ext_vector_type(4)))  int          v4i;

#define H_    12
#define D_    768
#define DK_   64
#define E_    4
#define EBD_  32
#define B_    8
#define N_    512
#define NREL  7
#define MTOK  4096                 // B_*N_
#define NEGF  (-1e9f)
#define SCALEF 0.125f              // DK^-0.5

__device__ __forceinline__ int rel_index(int a, int bk) {
  // relation table: 0:(1,1) 1:(1,0) 2:(2,2) 3:(2,0) 4:(0,0) 5:(0,1) 6:(0,2)
  if (a == 0) return 4 + bk;
  if (bk == a) return (a - 1) * 2;
  if (bk == 0) return (a - 1) * 2 + 1;
  return -1;
}

__device__ __forceinline__ v8f zero_v8f() {
  v8f z;
#pragma unroll
  for (int i = 0; i < 8; ++i) z[i] = 0.0f;
  return z;
}

// ---------------------------------------------------------------------------
// 1) Routing: top-2 softmax gates for each (relation, projection, head).
// ---------------------------------------------------------------------------
__global__ __launch_bounds__(256)
void route_kernel(const float* __restrict__ emb,
                  const float* __restrict__ rq_w, const float* __restrict__ rq_b,
                  const float* __restrict__ rk_w, const float* __restrict__ rk_b,
                  const float* __restrict__ rv_w, const float* __restrict__ rv_b,
                  float* __restrict__ gates) {
  int t = threadIdx.x;
  if (t >= NREL * 3 * H_) return;
  int h    = t % H_;
  int relp = t / H_;
  int rel  = relp / 3, p = relp % 3;
  const int rel_a[NREL] = {1, 1, 2, 2, 0, 0, 0};
  const int rel_b[NREL] = {1, 0, 2, 0, 0, 1, 2};
  int a = rel_a[rel], bb = rel_b[rel];
  const float* w  = (p == 0) ? rq_w : (p == 1) ? rk_w : rv_w;
  const float* bs = (p == 0) ? rq_b : (p == 1) ? rk_b : rv_b;
  float l[E_];
#pragma unroll
  for (int e = 0; e < E_; ++e) l[e] = bs[h * E_ + e];
  for (int c = 0; c < 2 * EBD_; ++c) {
    float fc = (c < EBD_) ? emb[a * EBD_ + c] : emb[bb * EBD_ + (c - EBD_)];
#pragma unroll
    for (int e = 0; e < E_; ++e) l[e] += fc * w[(h * 2 * EBD_ + c) * E_ + e];
  }
  float mx = l[0];
#pragma unroll
  for (int e = 1; e < E_; ++e) mx = fmaxf(mx, l[e]);
  float s = 0.f, d[E_];
#pragma unroll
  for (int e = 0; e < E_; ++e) { d[e] = __expf(l[e] - mx); s += d[e]; }
#pragma unroll
  for (int e = 0; e < E_; ++e) d[e] /= s;
  int i0 = 0;
#pragma unroll
  for (int e = 1; e < E_; ++e) if (d[e] > d[i0]) i0 = e;
  int i1 = (i0 == 0) ? 1 : 0;
#pragma unroll
  for (int e = 0; e < E_; ++e) if (e != i0 && d[e] > d[i1]) i1 = e;
  float renorm = 1.0f / (d[i0] + d[i1] + 1e-9f);
  float g[E_] = {0.f, 0.f, 0.f, 0.f};
  g[i0] = d[i0] * renorm;
  g[i1] = d[i1] * renorm;
#pragma unroll
  for (int e = 0; e < E_; ++e) gates[(relp * H_ + h) * E_ + e] = g[e];
}

// ---------------------------------------------------------------------------
// 2) f32 -> f16 convert (also used for proj_w: row-major proj_w IS the
//    transposed-B layout needed by the final GEMM, since out = o @ proj_w.T).
// ---------------------------------------------------------------------------
__global__ __launch_bounds__(256)
void cvt_f16_kernel(const float* __restrict__ src, _Float16* __restrict__ dst, int n) {
  int i = blockIdx.x * 256 + threadIdx.x;
  if (i < n) dst[i] = (_Float16)src[i];
}

// ---------------------------------------------------------------------------
// 3) Effective weights, TRANSPOSED B layout: Wb_t[relp][col=h*64+dk][d].
// ---------------------------------------------------------------------------
__global__ __launch_bounds__(256)
void mix_w_kernel(const float* __restrict__ gates,
                  const float* __restrict__ eq, const float* __restrict__ ek,
                  const float* __restrict__ ev, _Float16* __restrict__ Wb) {
  int i = blockIdx.x * 256 + threadIdx.x;
  const int per = D_ * (H_ * DK_);
  if (i >= NREL * 3 * per) return;
  int relp = i / per;
  int rem  = i % per;
  int col  = rem / D_;       // h*64+dk
  int d    = rem % D_;
  int h = col >> 6, dk = col & 63;
  int p = relp % 3;
  const float* Ex = (p == 0) ? eq : (p == 1) ? ek : ev;
  float acc = 0.f;
#pragma unroll
  for (int e = 0; e < E_; ++e) {
    float g = gates[(relp * H_ + h) * E_ + e];
    acc += g * Ex[(((h * E_ + e) * (size_t)D_) + d) * DK_ + dk];
  }
  Wb[i] = (_Float16)acc;
}

// ---------------------------------------------------------------------------
// 4) WMMA GEMM: C[4096,768] = A[4096,768](f16) x B[768,768](f16), B given
//    TRANSPOSED (Bt[n][k]) so B fragments are 2 x global b128 per lane.
//    A tile (64x32) staged into LDS by the Tensor Data Mover (wave 0).
//    mode 0: C f16 row-major   (Q/K projections)
//    mode 1: C f32 + bias      (final output projection)
//    mode 2: C f16 transposed  (V projections -> [col][tok], packed b128 store)
// ---------------------------------------------------------------------------
__global__ __launch_bounds__(256)
void gemm_kernel(const _Float16* __restrict__ A, const _Float16* __restrict__ Bt,
                 _Float16* __restrict__ Cf16, float* __restrict__ Cf32,
                 const float* __restrict__ bias, int mode) {
  __shared__ __align__(16) _Float16 As[64 * 32];
  int tid = threadIdx.x;
  int lane = tid & 31, wave = tid >> 5;
  int mblk = blockIdx.x * 64, nblk = blockIdx.y * 128;
  int mt = wave >> 1, nt = wave & 1;
  int m = lane & 15, hi = lane >> 4;
  v8f acc[4];
#pragma unroll
  for (int g = 0; g < 4; ++g) acc[g] = zero_v8f();
  unsigned lds_off = (unsigned)(unsigned long long)(size_t)&As[0];

  for (int kb = 0; kb < D_; kb += 32) {
    if (wave == 0) {
      // TDM descriptor: 2-D tile, 64 rows x 32 cols of 2-byte data, row
      // stride 768 elements, destination = As (compact 64x32).
      unsigned long long ga =
          (unsigned long long)(size_t)(A + (size_t)mblk * D_ + kb);
      v4u g0;
      g0[0] = 1u;                                   // count=1 (valid user D#)
      g0[1] = lds_off;                              // lds_addr
      g0[2] = (unsigned)(ga & 0xffffffffu);         // global_addr[31:0]
      g0[3] = (unsigned)((ga >> 32) & 0x01ffffffu)  // global_addr[56:32]
              | (2u << 30);                         // type=2 ("image")
      v8i g1;
      g1[0] = 0x00010000;            // data_size=1 -> 2 bytes/elem
      g1[1] = (int)(D_ << 16);       // tensor_dim0 (low16 @ bits 79:48)
      g1[2] = (int)(MTOK << 16);     // tensor_dim0 hi=0 | tensor_dim1 low16
      g1[3] = (int)(32 << 16);       // tensor_dim1 hi=0 | tile_dim0 = 32
      g1[4] = 64;                    // tile_dim1 = 64, tile_dim2 = 0 (2-D)
      g1[5] = D_;                    // tensor_dim0_stride low32 = 768
      g1[6] = 0;                     // stride hi | tensor_dim1_stride low16
      g1[7] = 0;
      v4i gz4;
      gz4[0] = 0; gz4[1] = 0; gz4[2] = 0; gz4[3] = 0;
      v8i gz8;
#pragma unroll
      for (int z = 0; z < 8; ++z) gz8[z] = 0;
      // 6-arg form on this toolchain: (g0, g1, g2, g3, g?, cpol)
      __builtin_amdgcn_tensor_load_to_lds(g0, g1, gz4, gz4, gz8, 0);
      __builtin_amdgcn_s_wait_tensorcnt(0);
    }
    __syncthreads();
    // A fragment (ISA 16x32 f16 layout): 2 x ds_load_b128
    union { uint4 q[2]; v16h h; } ua;
    ua.q[0] = *(const uint4*)&As[(mt * 16 + m) * 32 + hi * 8];
    ua.q[1] = *(const uint4*)&As[(mt * 16 + m) * 32 + 16 + hi * 8];
#pragma unroll
    for (int g = 0; g < 4; ++g) {
      // B fragment b[i] = B[k=(i<8 ? hi*8+i : 16+hi*8+i-8)][n] = Bt[n][k]: 2 x global b128
      int col = nblk + nt * 64 + g * 16 + m;
      const _Float16* bp = Bt + (size_t)col * D_ + kb + hi * 8;
      union { uint4 q[2]; v16h h; } ub;
      ub.q[0] = *(const uint4*)bp;
      ub.q[1] = *(const uint4*)(bp + 16);
      acc[g] = __builtin_amdgcn_wmma_f32_16x16x32_f16(false, ua.h, false, ub.h,
                                                      (short)0, acc[g], false, false);
    }
    if (kb + 32 < D_)  // cache hint for next B panel (global_prefetch_b8)
      __builtin_prefetch(Bt + (size_t)(nblk + (tid >> 1)) * D_ + kb + 32, 0, 1);
    __syncthreads();
  }
#pragma unroll
  for (int g = 0; g < 4; ++g) {
    int col = nblk + nt * 64 + g * 16 + m;
    if (mode == 2) {  // transposed f16 store: 8 rows contiguous -> one b128
      union { uint4 q; _Float16 hh[8]; } u;
#pragma unroll
      for (int v = 0; v < 8; ++v) u.hh[v] = (_Float16)acc[g][v];
      *(uint4*)&Cf16[(size_t)col * MTOK + mblk + mt * 16 + hi * 8] = u.q;
    } else {
#pragma unroll
      for (int v = 0; v < 8; ++v) {
        int row = mblk + mt * 16 + hi * 8 + v;
        if (mode == 0)
          Cf16[(size_t)row * D_ + col] = (_Float16)acc[g][v];
        else
          Cf32[(size_t)row * D_ + col] = acc[g][v] + bias[col];
      }
    }
  }
}

// ---------------------------------------------------------------------------
// 5) Attention: one wave per (batch, head, 16-query tile).
//    Q/K buffers row-major [tok][col]; V buffers transposed [col][tok].
//    Relation-decomposed scores + select, softmax, masked P_r x V^r.
//    Per-key-tile relation bitmasks skip inactive (r, tile) WMMA work.
// ---------------------------------------------------------------------------
__global__ __launch_bounds__(32)
void attn_kernel(const _Float16* __restrict__ Pbase, const int* __restrict__ assignment,
                 _Float16* __restrict__ obuf) {
  __shared__ float S[16 * N_];    // 32 KB score/prob buffer
  __shared__ int   ka[N_];
  __shared__ int   qa[16];
  __shared__ int   mask16[N_ / 16];  // active-relation bitmask per 16-key tile
  const size_t PSZ = (size_t)MTOK * (H_ * DK_);
  int lane = threadIdx.x;
  int qt = blockIdx.x, h = blockIdx.y, b = blockIdx.z;
  int q0 = qt * 16;
  for (int j = lane; j < N_; j += 32) ka[j] = assignment[b * N_ + j];
  if (lane < 16) qa[lane] = assignment[b * N_ + q0 + lane];
  for (int i = lane; i < 16 * N_; i += 32) S[i] = NEGF;
  __syncthreads();

  int qab = 0;
  for (int i = 0; i < 16; ++i) qab |= 1 << qa[i];
  {  // one 16-key tile per lane
    int mk = 0;
    for (int jj = 0; jj < 16; ++jj) {
      int bk = ka[lane * 16 + jj];
      for (int a = 0; a < 3; ++a)
        if ((qab >> a) & 1) {
          int r = rel_index(a, bk);
          if (r >= 0) mk |= 1 << r;
        }
    }
    mask16[lane] = mk;
  }
  __syncthreads();

  int m = lane & 15, hi = lane >> 4;

  // ---- Phase 1: relation-decomposed scores ----
  for (int r = 0; r < NREL; ++r) {
    const _Float16* Qb = Pbase + (size_t)(r * 3 + 0) * PSZ;
    const _Float16* Kb = Pbase + (size_t)(r * 3 + 1) * PSZ;
    union { uint4 q[2]; v16h h; } qf[2];
#pragma unroll
    for (int c = 0; c < 2; ++c) {  // DK=64 -> two K=32 chunks
      size_t rowoff = (size_t)(b * N_ + q0 + m) * (H_ * DK_);
      int colb = h * DK_ + c * 32 + hi * 8;
      qf[c].q[0] = *(const uint4*)&Qb[rowoff + colb];
      qf[c].q[1] = *(const uint4*)&Qb[rowoff + colb + 16];
    }
    for (int jt = 0; jt < N_ / 16; ++jt) {
      if (((mask16[jt] >> r) & 1) == 0) continue;  // wave-uniform skip
      v8f sc = zero_v8f();
#pragma unroll
      for (int c = 0; c < 2; ++c) {
        union { uint4 q[2]; v16h h; } kf;  // b[i] = K[key_n][dk=c*32+(i<8 ? hi*8+i : 16+hi*8+i-8)]
        size_t rowoff = (size_t)(b * N_ + jt * 16 + m) * (H_ * DK_);
        int colb = h * DK_ + c * 32 + hi * 8;
        kf.q[0] = *(const uint4*)&Kb[rowoff + colb];
        kf.q[1] = *(const uint4*)&Kb[rowoff + colb + 16];
        sc = __builtin_amdgcn_wmma_f32_16x16x32_f16(false, qf[c].h, false, kf.h,
                                                    (short)0, sc, false, false);
      }
      int j = jt * 16 + m;
      int bk = ka[j];
#pragma unroll
      for (int v = 0; v < 8; ++v) {
        int il = hi * 8 + v;
        if (rel_index(qa[il], bk) == r) S[il * N_ + j] = sc[v] * SCALEF;
      }
    }
  }
  __syncthreads();

  // ---- Phase 2: softmax (lanes 0..15, one row each) ----
  if (lane < 16) {
    float mx = -3.4e38f;
    for (int j = 0; j < N_; ++j) mx = fmaxf(mx, S[lane * N_ + j]);
    float sum = 0.f;
    for (int j = 0; j < N_; ++j) {
      float e = __expf(S[lane * N_ + j] - mx);
      S[lane * N_ + j] = e;
      sum += e;
    }
    float inv = 1.0f / sum;
    for (int j = 0; j < N_; ++j) S[lane * N_ + j] *= inv;
  }
  __syncthreads();

  // ---- Phase 3: O = sum_r P_r x V^r (V transposed: [col][tok]) ----
  int aq = qa[m];
  v8f oacc[4];
#pragma unroll
  for (int g = 0; g < 4; ++g) oacc[g] = zero_v8f();
  for (int r = 0; r < NREL; ++r) {
    const _Float16* Vt = Pbase + (size_t)(r * 3 + 2) * PSZ;
    for (int kc = 0; kc < N_ / 32; ++kc) {
      int mk = mask16[2 * kc] | mask16[2 * kc + 1];
      if (((mk >> r) & 1) == 0) continue;  // wave-uniform skip
      v16h pf;  // masked probabilities A fragment (16 queries x 32 keys)
#pragma unroll
      for (int i = 0; i < 8; ++i) {
        int j0 = kc * 32 + hi * 8 + i;
        int j1 = j0 + 16;
        pf[i]     = (rel_index(aq, ka[j0]) == r) ? (_Float16)S[m * N_ + j0] : (_Float16)0.f;
        pf[8 + i] = (rel_index(aq, ka[j1]) == r) ? (_Float16)S[m * N_ + j1] : (_Float16)0.f;
      }
#pragma unroll
      for (int g = 0; g < 4; ++g) {
        // b[i] = V[key=kc*32+(i<8 ? hi*8+i : 16+hi*8+i-8)][dk] = Vt[col][key]: 2 x global b128
        union { uint4 q[2]; v16h h; } vf;
        size_t base = (size_t)(h * DK_ + g * 16 + m) * MTOK
                    + (size_t)(b * N_ + kc * 32 + hi * 8);
        vf.q[0] = *(const uint4*)&Vt[base];
        vf.q[1] = *(const uint4*)&Vt[base + 16];
        oacc[g] = __builtin_amdgcn_wmma_f32_16x16x32_f16(false, pf, false, vf.h,
                                                         (short)0, oacc[g], false, false);
      }
    }
  }
#pragma unroll
  for (int g = 0; g < 4; ++g) {
#pragma unroll
    for (int v = 0; v < 8; ++v) {
      int row = b * N_ + q0 + hi * 8 + v;
      int col = h * DK_ + g * 16 + m;
      obuf[(size_t)row * (H_ * DK_) + col] = (_Float16)oacc[g][v];
    }
  }
}

// ---------------------------------------------------------------------------
// Launch
// ---------------------------------------------------------------------------
extern "C" void kernel_launch(void* const* d_in, const int* in_sizes, int n_in,
                              void* d_out, int out_size, void* d_ws, size_t ws_size,
                              hipStream_t stream) {
  (void)in_sizes; (void)n_in; (void)out_size; (void)ws_size;
  const float* x      = (const float*)d_in[0];
  const int*   assign = (const int*)d_in[1];
  const float* emb    = (const float*)d_in[2];
  const float* rq_w   = (const float*)d_in[3];
  const float* rq_b   = (const float*)d_in[4];
  const float* rk_w   = (const float*)d_in[5];
  const float* rk_b   = (const float*)d_in[6];
  const float* rv_w   = (const float*)d_in[7];
  const float* rv_b   = (const float*)d_in[8];
  const float* eq     = (const float*)d_in[9];
  const float* ek     = (const float*)d_in[10];
  const float* ev     = (const float*)d_in[11];
  const float* proj_w = (const float*)d_in[12];
  const float* proj_b = (const float*)d_in[13];

  char* w = (char*)d_ws;
  auto take = [&](size_t bytes) {
    char* p = w;
    w += (bytes + 255) & ~(size_t)255;
    return p;
  };
  float*    gates = (float*)take((size_t)NREL * 3 * H_ * E_ * sizeof(float));
  _Float16* xh    = (_Float16*)take((size_t)MTOK * D_ * 2);
  _Float16* Wb    = (_Float16*)take((size_t)NREL * 3 * D_ * D_ * 2);  // transposed
  _Float16* P     = (_Float16*)take((size_t)NREL * 3 * MTOK * D_ * 2);
  _Float16* obuf  = (_Float16*)take((size_t)MTOK * D_ * 2);
  _Float16* pwh   = (_Float16*)take((size_t)D_ * D_ * 2);  // proj_w as f16 (== B^T)

  route_kernel<<<1, 256, 0, stream>>>(emb, rq_w, rq_b, rk_w, rk_b, rv_w, rv_b, gates);
  cvt_f16_kernel<<<(MTOK * D_ + 255) / 256, 256, 0, stream>>>(x, xh, MTOK * D_);
  cvt_f16_kernel<<<(D_ * D_ + 255) / 256, 256, 0, stream>>>(proj_w, pwh, D_ * D_);
  mix_w_kernel<<<(NREL * 3 * D_ * D_ + 255) / 256, 256, 0, stream>>>(gates, eq, ek, ev, Wb);

  dim3 gg(MTOK / 64, D_ / 128);
  for (int rp = 0; rp < NREL * 3; ++rp) {
    int mode = (rp % 3 == 2) ? 2 : 0;  // V projections stored transposed
    gemm_kernel<<<gg, 256, 0, stream>>>(xh, Wb + (size_t)rp * D_ * D_,
                                        P + (size_t)rp * MTOK * D_,
                                        nullptr, nullptr, mode);
  }
  attn_kernel<<<dim3(N_ / 16, H_, B_), 32, 0, stream>>>(P, assign, obuf);
  gemm_kernel<<<gg, 256, 0, stream>>>(obuf, pwh, nullptr, (float*)d_out, proj_b, 1);
}
